// MPGCN_46669114638840
// MI455X (gfx1250) — matrix operand order repaired and checked by reading, not compiled, not run on hardware
//
#include <hip/hip_runtime.h>
#include <math.h>

// ---------------------------------------------------------------------------
// MPGCN step for MI455X (gfx1250, wave32, WMMA f32_16x16x32_f16)
//   N=50000, D=128, P=16, EH=NH=32, IN_S=36, IN_T=20, UPD=403, CTL=147
// Workspace: dh_space[N*128] f32, dh_time[N*128] f32 (51.2 MB),
//            then f16-packed weight fragments (~56 KB).
// ---------------------------------------------------------------------------

typedef __attribute__((ext_vector_type(16))) _Float16 v16h;
typedef __attribute__((ext_vector_type(8)))  float    v8f;

// ----- activation helpers (f32, match jax defaults) -----
__device__ __forceinline__ float gelu_f(float x) {
    const float c = 0.7978845608028654f;      // sqrt(2/pi)
    return 0.5f * x * (1.f + tanhf(c * (x + 0.044715f * x * x * x)));
}
__device__ __forceinline__ float silu_f(float x) {
    return x / (1.f + __expf(-x));
}

// ----- WMMA fragment builders (CDNA5 16-bit layouts, ISA 7.12.2) -----
// A (16xK tile, f16 in LDS, row-major, stride ldk). lane L holds row M=L%16;
// half h -> K = (h&7) + 8*(L/16) + 16*(h>>3)  (compiles to 2x ds_load_b128)
__device__ __forceinline__ v16h a_frag_lds(const _Float16* x, int ldk, int k0) {
    const int lane = threadIdx.x & 31;
    const _Float16* row = x + (lane & 15) * ldk;
    const int g = (lane >> 4) * 8;
    v16h a;
#pragma unroll
    for (int h = 0; h < 16; ++h) {
        int k = k0 + (h & 7) + g + ((h >> 3) << 4);
        a[h] = row[k];
    }
    return a;
}

// B fragment from pre-packed f16 buffer: layout [frag][lane][16 halves],
// 32 contiguous bytes per lane -> 2x global_load_b128, no conversion.
__device__ __forceinline__ v16h b_frag_packed(const _Float16* __restrict__ p,
                                              int frag) {
    const int lane = threadIdx.x & 31;
    return *(const v16h*)(p + ((size_t)frag * 32 + lane) * 16);
}

// D/C (16x16 f32): lane L holds col N=L%16, VGPR r holds row M = r + 8*(L/16)
__device__ __forceinline__ void d_store_lds(float* out, int ldn, int n0, v8f d) {
    const int lane = threadIdx.x & 31;
    float* col = out + n0 + (lane & 15);
    const int mb = (lane >> 4) * 8;
#pragma unroll
    for (int r = 0; r < 8; ++r) col[(size_t)(mb + r) * ldn] = d[r];
}

#define WMMA_F16(a, b, c) \
    __builtin_amdgcn_wmma_f32_16x16x32_f16(false, (a), false, (b), (short)0, (c), false, false)

// ---------------------------------------------------------------------------
// zero workspace accumulators
// ---------------------------------------------------------------------------
__global__ void zero_kernel(float* __restrict__ p, size_t n) {
    size_t i = (size_t)blockIdx.x * blockDim.x + threadIdx.x;
    size_t stride = (size_t)gridDim.x * blockDim.x;
    for (; i < n; i += stride) p[i] = 0.f;
}

// ---------------------------------------------------------------------------
// Pack W[nout][nin] (f32, row-major) into B-fragment order f16:
//   out[((nt*nkc + kc)*32 + lane)*16 + h], with
//   n = nt*16 + lane%16, k = kc*32 + 8*(lane/16) + (h&7) + 16*(h>>3), zero-pad k >= nin.
// ---------------------------------------------------------------------------
__global__ void pack_b_kernel(const float* __restrict__ W, int nin, int nkc,
                              _Float16* __restrict__ out, int total) {
    int idx = blockIdx.x * blockDim.x + threadIdx.x;
    if (idx >= total) return;
    const int h    = idx & 15;
    const int lane = (idx >> 4) & 31;
    const int f    = idx >> 9;
    const int kc   = f % nkc;
    const int nt   = f / nkc;
    const int k = kc * 32 + ((lane >> 4) << 3) + (h & 7) + ((h >> 3) << 4);
    const int n = nt * 16 + (lane & 15);
    out[idx] = (_Float16)((k < nin) ? W[(size_t)n * nin + k] : 0.f);
}

// ---------------------------------------------------------------------------
// Edge pipeline: gather -> (INP->32 WMMA) -> GELU -> LN -> dot(W2) -> scatter
//   128 edges / block, 256 threads (8 waves, 16 edges per wave)
// ---------------------------------------------------------------------------
template <int INP, int KPAD, bool TWO_DYN>
__global__ void __launch_bounds__(256)
edge_kernel(const float* __restrict__ wE, const float* __restrict__ tfeat,
            const float* __restrict__ h_dyn, const float* __restrict__ hmsg,
            const int* __restrict__ src, const int* __restrict__ dst, int nE,
            const _Float16* __restrict__ W1p, const float* __restrict__ b1,
            const float* __restrict__ ln_g, const float* __restrict__ ln_b,
            const float* __restrict__ W2, const float* __restrict__ b2,
            float* __restrict__ dh) {
    const int NKC = KPAD / 32;
    __shared__ _Float16 xs[128 * KPAD];   // edge features, f16
    __shared__ float    hid[128 * 32];    // hidden pre-activation, f32
    __shared__ float    fsh[128];         // per-edge gate f
    __shared__ int      ssrc[128], sdst[128];

    const int tid = threadIdx.x;
    const int e0  = blockIdx.x * 128;

    // ---- gather + build input rows (one edge per thread, f16 staging) ----
    if (tid < 128) {
        const int eg = e0 + tid;
        _Float16* row = xs + tid * KPAD;
        if (eg < nE) {
            const int s = src[eg], d = dst[eg];
            ssrc[tid] = s; sdst[tid] = d;
            row[0] = (_Float16)wE[eg];
#pragma unroll
            for (int j = 0; j < 3; ++j) row[1 + j] = (_Float16)tfeat[(size_t)d * 3 + j];
#pragma unroll
            for (int j = 0; j < 16; ++j) row[4 + j] = (_Float16)h_dyn[(size_t)s * 16 + j];
            if (TWO_DYN) {
#pragma unroll
                for (int j = 0; j < 16; ++j) row[20 + j] = (_Float16)h_dyn[(size_t)d * 16 + j];
            }
#pragma unroll
            for (int j = INP; j < KPAD; ++j) row[j] = (_Float16)0.f;
        } else {
            ssrc[tid] = 0; sdst[tid] = 0;
            for (int j = 0; j < KPAD; ++j) row[j] = (_Float16)0.f;
        }
    }
    __syncthreads();

    // ---- layer 1: [16 x INP] @ [INP x 32] via WMMA ----
    const int wave = tid >> 5;
    const _Float16* xrow = xs + (size_t)(wave * 16) * KPAD;
#pragma unroll
    for (int nt = 0; nt < 2; ++nt) {
        v8f acc = {};
#pragma unroll
        for (int kc = 0; kc < NKC; ++kc) {
            v16h a = a_frag_lds(xrow, KPAD, kc * 32);
            v16h b = b_frag_packed(W1p, nt * NKC + kc);
            acc = WMMA_F16(a, b, acc);
        }
        d_store_lds(hid + (size_t)(wave * 16) * 32, 32, nt * 16, acc);
    }
    __syncthreads();

    // ---- GELU -> LayerNorm -> dot(W2)+b2 (one edge per thread) ----
    if (tid < 128) {
        float v[32];
        float mu = 0.f;
#pragma unroll
        for (int j = 0; j < 32; ++j) {
            float z = gelu_f(hid[tid * 32 + j] + b1[j]);
            v[j] = z; mu += z;
        }
        mu *= (1.f / 32.f);
        float var = 0.f;
#pragma unroll
        for (int j = 0; j < 32; ++j) { float d = v[j] - mu; var += d * d; }
        var *= (1.f / 32.f);
        const float inv = rsqrtf(var + 1e-5f);
        float f = b2[0];
#pragma unroll
        for (int j = 0; j < 32; ++j)
            f += ((v[j] - mu) * inv * ln_g[j] + ln_b[j]) * W2[j];
        fsh[tid] = f;
    }
    __syncthreads();

    // ---- scatter: dh[dst] += h[src] * f  (128-lane coalesced rows) ----
    for (int idx = tid; idx < 128 * 128; idx += 256) {
        const int e = idx >> 7, c = idx & 127;
        if (e0 + e < nE) {
            float m = hmsg[(size_t)ssrc[e] * 128 + c] * fsh[e];
            atomicAdd(dh + (size_t)sdst[e] * 128 + c, m);
        }
    }
}

// ---------------------------------------------------------------------------
// Node update MLP: [h|dh_s|dh_t|h_dyn|tfeat](403) ->32 GELU ->32 LN GELU ->128
//   64 nodes / block, 128 threads (4 waves, 16 nodes per wave)
// ---------------------------------------------------------------------------
__global__ void __launch_bounds__(128)
node_kernel(const float* __restrict__ h, const float* __restrict__ dh_s,
            const float* __restrict__ dh_t, const float* __restrict__ h_dyn,
            const float* __restrict__ tfeat, int nN,
            const _Float16* __restrict__ W1p, const float* __restrict__ b1,
            const _Float16* __restrict__ W2p, const float* __restrict__ b2,
            const float* __restrict__ g,  const float* __restrict__ be,
            const _Float16* __restrict__ W3p, const float* __restrict__ b3,
            float* __restrict__ out /* [N,144] */) {
    const int KP = 416;                    // 403 padded to 13*32
    __shared__ _Float16 xs[64 * 416];      // 53 KB
    __shared__ float    buf[64 * 32];
    __shared__ _Float16 zh[64 * 32];

    const int tid = threadIdx.x;
    const int n0  = blockIdx.x * 64;

    // ---- build input tile ----
    for (int idx = tid; idx < 64 * KP; idx += 128) {
        const int r = idx / KP, c = idx % KP;
        const int node = n0 + r;
        float v = 0.f;
        if (node < nN) {
            if      (c < 128) v = h   [(size_t)node * 128 + c];
            else if (c < 256) v = dh_s[(size_t)node * 128 + (c - 128)];
            else if (c < 384) v = dh_t[(size_t)node * 128 + (c - 256)];
            else if (c < 400) v = h_dyn[(size_t)node * 16 + (c - 384)];
            else if (c < 403) v = tfeat[(size_t)node * 3 + (c - 400)];
        }
        xs[idx] = (_Float16)v;
    }
    __syncthreads();

    const int wave = tid >> 5;
    const _Float16* xrow = xs + (size_t)(wave * 16) * KP;
    float* hrow = buf + (size_t)(wave * 16) * 32;

    // ---- layer 1: 403 -> 32 ----
#pragma unroll
    for (int nt = 0; nt < 2; ++nt) {
        v8f acc = {};
#pragma unroll
        for (int kc = 0; kc < 13; ++kc) {
            v16h a = a_frag_lds(xrow, KP, kc * 32);
            v16h b = b_frag_packed(W1p, nt * 13 + kc);
            acc = WMMA_F16(a, b, acc);
        }
        d_store_lds(hrow, 32, nt * 16, acc);
    }
    __syncthreads();

    // ---- z1 = gelu(. + b1), restage as f16 ----
    for (int idx = tid; idx < 64 * 32; idx += 128)
        zh[idx] = (_Float16)gelu_f(buf[idx] + b1[idx & 31]);
    __syncthreads();

    // ---- layer 2: 32 -> 32 ----
    const _Float16* zrow = zh + (size_t)(wave * 16) * 32;
    {
        v16h a = a_frag_lds(zrow, 32, 0);
#pragma unroll
        for (int nt = 0; nt < 2; ++nt) {
            v16h b = b_frag_packed(W2p, nt);
            v8f acc = {};
            acc = WMMA_F16(a, b, acc);
            d_store_lds(hrow, 32, nt * 16, acc);
        }
    }
    __syncthreads();

    // ---- z2 = gelu(ln(. + b2)) per node ----
    if (tid < 64) {
        float v[32];
        float mu = 0.f;
#pragma unroll
        for (int j = 0; j < 32; ++j) { v[j] = buf[tid * 32 + j] + b2[j]; mu += v[j]; }
        mu *= (1.f / 32.f);
        float var = 0.f;
#pragma unroll
        for (int j = 0; j < 32; ++j) { float d = v[j] - mu; var += d * d; }
        var *= (1.f / 32.f);
        const float inv = rsqrtf(var + 1e-5f);
#pragma unroll
        for (int j = 0; j < 32; ++j)
            zh[tid * 32 + j] = (_Float16)gelu_f((v[j] - mu) * inv * g[j] + be[j]);
    }
    __syncthreads();

    // ---- layer 3: 32 -> 128, write d_out cols [0,128) ----
    {
        v16h a = a_frag_lds(zrow, 32, 0);
        const int lane = tid & 31;
        const int mb = (lane >> 4) * 8;
#pragma unroll
        for (int nt = 0; nt < 8; ++nt) {
            v16h b = b_frag_packed(W3p, nt);
            v8f acc = {};
            acc = WMMA_F16(a, b, acc);
            const int col = nt * 16 + (lane & 15);
#pragma unroll
            for (int r = 0; r < 8; ++r) {
                const int node = n0 + wave * 16 + mb + r;
                if (node < nN) out[(size_t)node * 144 + col] = acc[r] + b3[col];
            }
        }
    }
}

// ---------------------------------------------------------------------------
// Control MLP: [h|h_dyn|tfeat](147) ->32 SiLU ->32 LN SiLU ->16
// ---------------------------------------------------------------------------
__global__ void __launch_bounds__(128)
ctrl_kernel(const float* __restrict__ h, const float* __restrict__ h_dyn,
            const float* __restrict__ tfeat, int nN,
            const _Float16* __restrict__ W1p, const float* __restrict__ b1,
            const _Float16* __restrict__ W2p, const float* __restrict__ b2,
            const float* __restrict__ g,  const float* __restrict__ be,
            const _Float16* __restrict__ W3p, const float* __restrict__ b3,
            float* __restrict__ out /* [N,144] */) {
    const int KP = 160;                    // 147 padded to 5*32
    __shared__ _Float16 xs[64 * 160];
    __shared__ float    buf[64 * 32];
    __shared__ _Float16 zh[64 * 32];

    const int tid = threadIdx.x;
    const int n0  = blockIdx.x * 64;

    for (int idx = tid; idx < 64 * KP; idx += 128) {
        const int r = idx / KP, c = idx % KP;
        const int node = n0 + r;
        float v = 0.f;
        if (node < nN) {
            if      (c < 128) v = h[(size_t)node * 128 + c];
            else if (c < 144) v = h_dyn[(size_t)node * 16 + (c - 128)];
            else if (c < 147) v = tfeat[(size_t)node * 3 + (c - 144)];
        }
        xs[idx] = (_Float16)v;
    }
    __syncthreads();

    const int wave = tid >> 5;
    const _Float16* xrow = xs + (size_t)(wave * 16) * KP;
    float* hrow = buf + (size_t)(wave * 16) * 32;

#pragma unroll
    for (int nt = 0; nt < 2; ++nt) {
        v8f acc = {};
#pragma unroll
        for (int kc = 0; kc < 5; ++kc) {
            v16h a = a_frag_lds(xrow, KP, kc * 32);
            v16h b = b_frag_packed(W1p, nt * 5 + kc);
            acc = WMMA_F16(a, b, acc);
        }
        d_store_lds(hrow, 32, nt * 16, acc);
    }
    __syncthreads();

    for (int idx = tid; idx < 64 * 32; idx += 128)
        zh[idx] = (_Float16)silu_f(buf[idx] + b1[idx & 31]);
    __syncthreads();

    const _Float16* zrow = zh + (size_t)(wave * 16) * 32;
    {
        v16h a = a_frag_lds(zrow, 32, 0);
#pragma unroll
        for (int nt = 0; nt < 2; ++nt) {
            v16h b = b_frag_packed(W2p, nt);
            v8f acc = {};
            acc = WMMA_F16(a, b, acc);
            d_store_lds(hrow, 32, nt * 16, acc);
        }
    }
    __syncthreads();

    if (tid < 64) {
        float v[32];
        float mu = 0.f;
#pragma unroll
        for (int j = 0; j < 32; ++j) { v[j] = buf[tid * 32 + j] + b2[j]; mu += v[j]; }
        mu *= (1.f / 32.f);
        float var = 0.f;
#pragma unroll
        for (int j = 0; j < 32; ++j) { float d = v[j] - mu; var += d * d; }
        var *= (1.f / 32.f);
        const float inv = rsqrtf(var + 1e-5f);
#pragma unroll
        for (int j = 0; j < 32; ++j)
            zh[tid * 32 + j] = (_Float16)silu_f((v[j] - mu) * inv * g[j] + be[j]);
    }
    __syncthreads();

    // layer 3: 32 -> 16, write d_out cols [128,144)
    {
        v16h a = a_frag_lds(zrow, 32, 0);
        v16h b = b_frag_packed(W3p, 0);
        v8f acc = {};
        acc = WMMA_F16(a, b, acc);
        const int lane = tid & 31;
        const int mb = (lane >> 4) * 8;
        const int col = lane & 15;
#pragma unroll
        for (int r = 0; r < 8; ++r) {
            const int node = n0 + wave * 16 + mb + r;
            if (node < nN) out[(size_t)node * 144 + 128 + col] = acc[r] + b3[col];
        }
    }
}

// ---------------------------------------------------------------------------
extern "C" void kernel_launch(void* const* d_in, const int* in_sizes, int n_in,
                              void* d_out, int out_size, void* d_ws, size_t ws_size,
                              hipStream_t stream) {
    (void)n_in; (void)out_size; (void)ws_size;
    const float* h        = (const float*)d_in[0];
    const float* h_prev   = (const float*)d_in[1];
    const float* h_dyn    = (const float*)d_in[2];
    const float* tfeat    = (const float*)d_in[3];
    const float* w_space  = (const float*)d_in[4];
    const float* w_time   = (const float*)d_in[5];
    const int*   src_s    = (const int*)d_in[6];
    const int*   dst_s    = (const int*)d_in[7];
    const int*   src_t    = (const int*)d_in[8];
    const int*   dst_t    = (const int*)d_in[9];
    // params in dict order
    const float* se_W1 = (const float*)d_in[10];
    const float* se_b1 = (const float*)d_in[11];
    const float* se_g  = (const float*)d_in[12];
    const float* se_be = (const float*)d_in[13];
    const float* se_W2 = (const float*)d_in[14];
    const float* se_b2 = (const float*)d_in[15];
    const float* te_W1 = (const float*)d_in[16];
    const float* te_b1 = (const float*)d_in[17];
    const float* te_g  = (const float*)d_in[18];
    const float* te_be = (const float*)d_in[19];
    const float* te_W2 = (const float*)d_in[20];
    const float* te_b2 = (const float*)d_in[21];
    const float* nd_W1 = (const float*)d_in[22];
    const float* nd_b1 = (const float*)d_in[23];
    const float* nd_W2 = (const float*)d_in[24];
    const float* nd_b2 = (const float*)d_in[25];
    const float* nd_g  = (const float*)d_in[26];
    const float* nd_be = (const float*)d_in[27];
    const float* nd_W3 = (const float*)d_in[28];
    const float* nd_b3 = (const float*)d_in[29];
    const float* cu_W1 = (const float*)d_in[30];
    const float* cu_b1 = (const float*)d_in[31];
    const float* cu_W2 = (const float*)d_in[32];
    const float* cu_b2 = (const float*)d_in[33];
    const float* cu_g  = (const float*)d_in[34];
    const float* cu_be = (const float*)d_in[35];
    const float* cu_W3 = (const float*)d_in[36];
    const float* cu_b3 = (const float*)d_in[37];

    const int N  = in_sizes[0] / 128;  // 50000
    const int ES = in_sizes[6];        // 800000
    const int ET = in_sizes[8];        // 50000

    float* dh_space = (float*)d_ws;
    float* dh_time  = dh_space + (size_t)N * 128;

    // packed f16 weight-fragment region (after the two accumulators; the
    // accumulator region is 32-byte aligned so v16h loads are aligned)
    _Float16* pw = (_Float16*)(dh_time + (size_t)N * 128);
    const int F = 512;  // halves per fragment (32 lanes x 16)
    _Float16* se_W1p = pw;               // 2nt x 2kc = 4 frags
    _Float16* te_W1p = se_W1p + 4 * F;   // 2nt x 1kc = 2
    _Float16* nd_W1p = te_W1p + 2 * F;   // 2nt x 13kc = 26
    _Float16* nd_W2p = nd_W1p + 26 * F;  // 2
    _Float16* nd_W3p = nd_W2p + 2 * F;   // 8nt x 1kc = 8
    _Float16* cu_W1p = nd_W3p + 8 * F;   // 2nt x 5kc = 10
    _Float16* cu_W2p = cu_W1p + 10 * F;  // 2
    _Float16* cu_W3p = cu_W2p + 2 * F;   // 1

    // zero both accumulators (atomics accumulate into these each call)
    const size_t zn = (size_t)N * 256;
    zero_kernel<<<2048, 256, 0, stream>>>(dh_space, zn);

    // pack weights into B-fragment order (tiny, once per launch)
    pack_b_kernel<<<(4 * F + 255) / 256, 256, 0, stream>>>(se_W1, 36, 2, se_W1p, 4 * F);
    pack_b_kernel<<<(2 * F + 255) / 256, 256, 0, stream>>>(te_W1, 20, 1, te_W1p, 2 * F);
    pack_b_kernel<<<(26 * F + 255) / 256, 256, 0, stream>>>(nd_W1, 403, 13, nd_W1p, 26 * F);
    pack_b_kernel<<<(2 * F + 255) / 256, 256, 0, stream>>>(nd_W2, 32, 1, nd_W2p, 2 * F);
    pack_b_kernel<<<(8 * F + 255) / 256, 256, 0, stream>>>(nd_W3, 32, 1, nd_W3p, 8 * F);
    pack_b_kernel<<<(10 * F + 255) / 256, 256, 0, stream>>>(cu_W1, 147, 5, cu_W1p, 10 * F);
    pack_b_kernel<<<(2 * F + 255) / 256, 256, 0, stream>>>(cu_W2, 32, 1, cu_W2p, 2 * F);
    pack_b_kernel<<<(1 * F + 255) / 256, 256, 0, stream>>>(cu_W3, 32, 1, cu_W3p, 1 * F);

    // edge pipelines (space: 36 in, dyn of src+dst; time: 20 in, dyn of src)
    edge_kernel<36, 64, true><<<(ES + 127) / 128, 256, 0, stream>>>(
        w_space, tfeat, h_dyn, h, src_s, dst_s, ES,
        se_W1p, se_b1, se_g, se_be, se_W2, se_b2, dh_space);
    edge_kernel<20, 32, false><<<(ET + 127) / 128, 256, 0, stream>>>(
        w_time, tfeat, h_dyn, h_prev, src_t, dst_t, ET,
        te_W1p, te_b1, te_g, te_be, te_W2, te_b2, dh_time);

    // node + control MLPs
    node_kernel<<<(N + 63) / 64, 128, 0, stream>>>(
        h, dh_space, dh_time, h_dyn, tfeat, N,
        nd_W1p, nd_b1, nd_W2p, nd_b2, nd_g, nd_be, nd_W3p, nd_b3, (float*)d_out);
    ctrl_kernel<<<(N + 63) / 64, 128, 0, stream>>>(
        h, h_dyn, tfeat, N,
        cu_W1p, cu_b1, cu_W2p, cu_b2, cu_g, cu_be, cu_W3p, cu_b3, (float*)d_out);
}
